// ParametricClampedBeam_56083682951607
// MI455X (gfx1250) — compile-verified, matrix operand order checked by reading, not run on hardware
//
#include <hip/hip_runtime.h>
#include <hip/hip_bf16.h>
#include <math.h>

typedef _Float16 v16h __attribute__((ext_vector_type(16)));
typedef _Float16 v8h  __attribute__((ext_vector_type(8)));
typedef float    v8f  __attribute__((ext_vector_type(8)));

#define E_REF 200000000000.0f
#define I_REF 1e-06f
#define Q_REF 1000.0f

#define WSTR 72      // LDS row stride in halves (144 B = 128 B data + 16 B pad; 16B-aligned, conflict-free)
#define NPB  128     // points per block (8 waves x 16 rows)

__device__ __forceinline__ v16h cat8(v8h lo, v8h hi) {
    return __builtin_shufflevector(lo, hi, 0,1,2,3,4,5,6,7,8,9,10,11,12,13,14,15);
}

// Branchless tanh: 1 - 2/(exp2(2*log2e*u)+1). Saturates to +-1 in f32 at large |u|.
// 3 VALU + v_exp_f32 + v_rcp_f32 vs OCML tanhf's branchy polynomial path.
__device__ __forceinline__ float fast_tanh(float u) {
    float e = __builtin_amdgcn_exp2f(u * 2.8853900817779268f);
    return 1.0f - 2.0f * __builtin_amdgcn_rcpf(e + 1.0f);
}

// One 64->64 layer for all 5 jet streams: D = H_s @ W + (s==0 ? b : 0), then tanh jet.
// wT: LDS f16 W^T[n][k] rows of stride WSTR. hst: per-wave LDS f16 staging tile [m][k].
__device__ __forceinline__ void mlp_layer(v8f (&h)[5][4], const _Float16* wT,
                                          const float* bias, _Float16* hst,
                                          int hl, int jl)
{
    #pragma unroll
    for (int s = 0; s < 5; ++s) {
        // Stage old h_s (C-layout f32 regs) into LDS as f16 [m][k]
        #pragma unroll
        for (int t = 0; t < 4; ++t)
            #pragma unroll
            for (int v = 0; v < 8; ++v)
                hst[(v + 8 * hl) * WSTR + (jl + 16 * t)] = (_Float16)h[s][t][v];

        // A operands: lane (hl, m=jl); element e -> K = e + 8*hl (+16 for e>=8), per ISA layout.
        v16h A[2];
        #pragma unroll
        for (int c = 0; c < 2; ++c) {
            const v8h* lo = (const v8h*)&hst[jl * WSTR + c * 32 + 8 * hl];
            const v8h* hi = (const v8h*)&hst[jl * WSTR + c * 32 + 8 * hl + 16];
            A[c] = cat8(*lo, *hi);
        }

        // GEMM: 4 N-tiles x 2 K-chunks of v_wmma_f32_16x16x32_f16
        #pragma unroll
        for (int t = 0; t < 4; ++t) {
            v8f acc;
            if (s == 0) {
                float b0 = bias[jl + 16 * t];
                #pragma unroll
                for (int v = 0; v < 8; ++v) acc[v] = b0;
            } else {
                acc = (v8f)(0.0f);   // zero C: candidate for inline SRC2=0 encoding
            }
            #pragma unroll
            for (int c = 0; c < 2; ++c) {
                const v8h* lo = (const v8h*)&wT[(jl + 16 * t) * WSTR + c * 32 + 8 * hl];
                const v8h* hi = (const v8h*)&wT[(jl + 16 * t) * WSTR + c * 32 + 8 * hl + 16];
                v16h B = cat8(*lo, *hi);
                acc = __builtin_amdgcn_wmma_f32_16x16x32_f16(false, A[c], false, B,
                                                             (short)0, acc, false, false);
            }
            h[s][t] = acc;
        }
        // Fence scheduling: keep the next stream's staging / the jet from being
        // interleaved into this GEMM region (controls VGPR pressure < 256).
        __builtin_amdgcn_sched_barrier(0);
    }

    // tanh jet (Faa di Bruno up to 4th order), elementwise in C layout, f32 VALU
    #pragma unroll
    for (int t = 0; t < 4; ++t)
        #pragma unroll
        for (int v = 0; v < 8; ++v) {
            float u  = h[0][t][v];
            float a1 = h[1][t][v], a2 = h[2][t][v], a3 = h[3][t][v], a4 = h[4][t][v];
            float th = fast_tanh(u);
            float t1 = 1.0f - th * th;
            float t2 = -2.0f * th * t1;
            float t3 = -2.0f * t1 * t1 - 2.0f * th * t2;
            float t4 = -6.0f * t1 * t2 - 2.0f * th * t3;
            h[0][t][v] = th;
            h[1][t][v] = t1 * a1;
            h[2][t][v] = t2 * a1 * a1 + t1 * a2;
            h[3][t][v] = t3 * a1 * a1 * a1 + 3.0f * t2 * a1 * a2 + t1 * a3;
            h[4][t][v] = t4 * a1 * a1 * a1 * a1 + 6.0f * t3 * a1 * a1 * a2
                       + t2 * (3.0f * a2 * a2 + 4.0f * a1 * a3) + t1 * a4;
        }
    __builtin_amdgcn_sched_barrier(0);
}

__global__ __launch_bounds__(256)
void ParametricClampedBeam_jet_wmma(const float* __restrict__ x, const float* __restrict__ E,
                                    const float* __restrict__ I, const float* __restrict__ q,
                                    const float* __restrict__ W1, const float* __restrict__ b1,
                                    const float* __restrict__ W2, const float* __restrict__ b2,
                                    const float* __restrict__ W3, const float* __restrict__ b3,
                                    const float* __restrict__ W4, const float* __restrict__ b4,
                                    float* __restrict__ out, int N)
{
    __shared__ __align__(16) _Float16 sW2T[64 * WSTR];
    __shared__ __align__(16) _Float16 sW3T[64 * WSTR];
    __shared__ __align__(16) _Float16 sHst[8][16 * WSTR];
    __shared__ float sW1[256];
    __shared__ float sB1[64], sB2[64], sB3[64], sW4v[64];
    __shared__ float sX[NPB], sF1[NPB], sF2[NPB], sF3[NPB], sS[NPB];

    const int tid = threadIdx.x;

    // Stage weights: W2^T, W3^T as f16 (B-operand source), small arrays as f32.
    for (int i = tid; i < 64 * 64; i += 256) {
        int k = i >> 6, n = i & 63;               // W[k][n] row-major (fin, fout)
        sW2T[n * WSTR + k] = (_Float16)W2[i];
        sW3T[n * WSTR + k] = (_Float16)W3[i];
    }
    sW1[tid] = W1[tid & 255];                      // 256 threads cover 4x64 exactly
    if (tid < 64) { sB1[tid] = b1[tid]; sB2[tid] = b2[tid]; sB3[tid] = b3[tid]; sW4v[tid] = W4[tid]; }
    if (tid < NPB) {
        long p = (long)blockIdx.x * NPB + tid;
        if (p >= N) p = N - 1;
        float xv = x[p], Ev = E[p], Iv = I[p], qv = q[p];
        sX[tid]  = xv;
        sF1[tid] = Ev * (1.0f / E_REF);
        sF2[tid] = Iv * (1.0f / I_REF);
        sF3[tid] = qv * (1.0f / Q_REF);
        sS[tid]  = qv / (Ev * Iv + 1.1920929e-07f);   // s = q L^4 / (E I + eps), L = 1
    }
    __syncthreads();

    const int lane = tid & 31;
    const int wave = tid >> 5;
    const int hl = lane >> 4;   // 16-lane half
    const int jl = lane & 15;   // N-col (C/B) or M-row (A) index within half
    _Float16* hst = sHst[wave];

    v8f h[5][4];                // 5 jet streams x 4 N-tiles, f32 C-layout

    // Layer 1 (4 -> 64). Input jet: d(inp)/dx = [1,0,0,0], higher derivs zero.
    #pragma unroll
    for (int t = 0; t < 4; ++t) {
        int j = jl + 16 * t;
        float w0 = sW1[0 * 64 + j], w1 = sW1[1 * 64 + j];
        float w2 = sW1[2 * 64 + j], w3 = sW1[3 * 64 + j];
        float bj = sB1[j];
        #pragma unroll
        for (int v = 0; v < 8; ++v) {
            int r = wave * 16 + v + 8 * hl;
            float u = sX[r] * w0 + sF1[r] * w1 + sF2[r] * w2 + sF3[r] * w3 + bj;
            float th = fast_tanh(u);
            float t1 = 1.0f - th * th;
            float t2 = -2.0f * th * t1;
            float t3 = -2.0f * t1 * t1 - 2.0f * th * t2;
            float t4 = -6.0f * t1 * t2 - 2.0f * th * t3;
            float ww = w0 * w0;
            h[0][t][v] = th;
            h[1][t][v] = t1 * w0;
            h[2][t][v] = t2 * ww;
            h[3][t][v] = t3 * ww * w0;
            h[4][t][v] = t4 * ww * ww;
        }
    }
    __builtin_amdgcn_sched_barrier(0);

    // Layers 2, 3: WMMA GEMMs + tanh jet
    mlp_layer(h, sW2T, sB2, hst, hl, jl);
    mlp_layer(h, sW3T, sB3, hst, hl, jl);

    // Layer 4 (64 -> 1) per stream: dot with W4 across tiles, butterfly-reduce across 16 lanes.
    float w4j[4];
    #pragma unroll
    for (int t = 0; t < 4; ++t) w4j[t] = sW4v[jl + 16 * t];

    float outv[5];
    #pragma unroll
    for (int s = 0; s < 5; ++s) {
        float p[8];
        #pragma unroll
        for (int v = 0; v < 8; ++v)
            p[v] = h[s][0][v] * w4j[0] + h[s][1][v] * w4j[1]
                 + h[s][2][v] * w4j[2] + h[s][3][v] * w4j[3];
        #pragma unroll
        for (int off = 1; off <= 8; off <<= 1)
            #pragma unroll
            for (int v = 0; v < 8; ++v)
                p[v] += __shfl_xor(p[v], off, 32);
        float sel = 0.0f;
        #pragma unroll
        for (int v = 0; v < 8; ++v) sel = (jl == v) ? p[v] : sel;
        outv[s] = sel;
    }

    // Lanes with jl<8 own one point each: row m = 8*hl + jl.
    if (jl < 8) {
        int m = 8 * hl + jl;
        int r = wave * 16 + m;
        long pt = (long)blockIdx.x * NPB + r;
        if (pt < N) {
            float psi = outv[0] + b4[0];
            float px = outv[1], pxx = outv[2], pxxx = outv[3], pxxxx = outv[4];
            float xm = sX[r], sm = sS[r], x2 = xm * xm;
            out[0L * N + pt] = sm * (x2 * psi);
            out[1L * N + pt] = sm * (2.0f * xm * psi + x2 * px);
            out[2L * N + pt] = sm * (2.0f * psi + 4.0f * xm * px + x2 * pxx);
            out[3L * N + pt] = sm * (6.0f * px + 6.0f * xm * pxx + x2 * pxxx);
            out[4L * N + pt] = sm * (12.0f * pxx + 8.0f * xm * pxxx + x2 * pxxxx);
        }
    }
}

extern "C" void kernel_launch(void* const* d_in, const int* in_sizes, int n_in,
                              void* d_out, int out_size, void* d_ws, size_t ws_size,
                              hipStream_t stream) {
    const float* x  = (const float*)d_in[0];
    const float* E  = (const float*)d_in[1];
    const float* I  = (const float*)d_in[2];
    const float* q  = (const float*)d_in[3];
    const float* W1 = (const float*)d_in[4];
    const float* b1 = (const float*)d_in[5];
    const float* W2 = (const float*)d_in[6];
    const float* b2 = (const float*)d_in[7];
    const float* W3 = (const float*)d_in[8];
    const float* b3 = (const float*)d_in[9];
    const float* W4 = (const float*)d_in[10];
    const float* b4 = (const float*)d_in[11];
    float* out = (float*)d_out;
    int N = in_sizes[0];
    int blocks = (N + NPB - 1) / NPB;   // 262144/128 = 2048
    ParametricClampedBeam_jet_wmma<<<blocks, 256, 0, stream>>>(
        x, E, I, q, W1, b1, W2, b2, W3, b3, W4, b4, out, N);
}
